// BinaryConv2d_17308718202988
// MI455X (gfx1250) — matrix operand order repaired and checked by reading, not compiled, not run on hardware
//
#include <hip/hip_runtime.h>

typedef _Float16 hf;
typedef __attribute__((ext_vector_type(16))) _Float16 v16h;
typedef __attribute__((ext_vector_type(8)))  _Float16 v8h;
typedef __attribute__((ext_vector_type(8)))  float    v8f;

#define NB   32
#define CIN  256
#define COUT 256
#define H    56
#define W    56
#define HP   58          // padded (pad=1 each side)
#define WP   58
#define PIX  (NB*H*W)    // 100352, divisible by 64

// ---------------------------------------------------------------------------
// Prolog 1: binarize weights f32 OIHW -> f16 [tap][co][c] (tap = kh*3+kw)
// ---------------------------------------------------------------------------
__global__ void binarize_w(const float* __restrict__ w, hf* __restrict__ wb) {
    int idx = blockIdx.x * blockDim.x + threadIdx.x;   // exactly 256*256*9 threads
    int co  = idx / (CIN * 9);
    int rem = idx - co * (CIN * 9);
    int c   = rem / 9;
    int tap = rem - c * 9;
    float v = w[idx];                                  // OIHW flat == co*2304 + c*9 + tap
    wb[((size_t)tap * COUT + co) * CIN + c] = (v >= 0.0f) ? (hf)1.0f : (hf)(-1.0f);
}

// ---------------------------------------------------------------------------
// Prolog 2: x f32 NCHW -> f16 padded NHWC [n][hp][wp][c], pad value = -1.0
// ---------------------------------------------------------------------------
__global__ void pad_nhwc(const float* __restrict__ x, hf* __restrict__ xh) {
    int idx = blockIdx.x * blockDim.x + threadIdx.x;   // exactly 32*58*58*256 threads
    int c  = idx & (CIN - 1);
    int t  = idx >> 8;
    int wp = t % WP;  t /= WP;
    int hp = t % HP;
    int n  = t / HP;
    float v = -1.0f;
    if (hp >= 1 && hp <= H && wp >= 1 && wp <= W)
        v = x[(((size_t)n * CIN + c) * H + (hp - 1)) * W + (wp - 1)];
    xh[idx] = (hf)v;
}

// ---------------------------------------------------------------------------
// Main: implicit GEMM. One wave32 owns a 32(co) x 64(pixel) output tile:
// 2 A-frags x 4 B-frags -> 8 f32 accumulators, 8 WMMAs per K=32 chunk.
// The 8 waves of a block cover co=0..255 for the SAME 64 pixels, so B
// fragments are fetched once per WGP and reused 8x.
// ---------------------------------------------------------------------------
__global__ void __launch_bounds__(256)
bconv_wmma(const hf* __restrict__ xh, const hf* __restrict__ wb,
           float* __restrict__ out) {
    const int lane   = threadIdx.x & 31;
    const int wave   = threadIdx.x >> 5;       // co-tile 0..7 (32 co each)
    const int l15    = lane & 15;
    const int hi     = lane >> 4;
    const int cobase = wave * 32;
    const int pbase  = blockIdx.x * 64;        // 64 pixels per block

    // A fragment rows (K order used here, same as B): lane l -> row co, K at +hi*16 (contiguous 16)
    const hf* arow0 = wb + (size_t)(cobase + l15) * CIN + hi * 16;
    const hf* arow1 = arow0 + (size_t)16 * CIN;

    // B fragment pixels (K order used here, same as A): lane l -> column pixel, K at +hi*16
    int nn[4], hh[4], ww[4];
    const hf* bpix[4];
#pragma unroll
    for (int ni = 0; ni < 4; ++ni) {
        int p = pbase + ni * 16 + l15;
        int n = p / (H * W);
        int r = p - n * (H * W);
        int h = r / W;
        int w = r - h * W;
        nn[ni] = n; hh[ni] = h; ww[ni] = w;
        // padded coords for tap (kh,kw) are (h+kh, w+kw) since pad==1
        bpix[ni] = xh + (((size_t)n * HP + h) * WP + w) * CIN + hi * 16;
    }

    v8f acc[2][4] = {};

#pragma unroll 1
    for (int tap = 0; tap < 9; ++tap) {
        const int kh = tap / 3;
        const int kw = tap - kh * 3;
        const hf* A0 = arow0 + (size_t)tap * (COUT * CIN);
        const hf* A1 = arow1 + (size_t)tap * (COUT * CIN);
        const size_t boff = ((size_t)kh * WP + kw) * CIN;
#pragma unroll
        for (int c0 = 0; c0 < CIN; c0 += 32) {
            union { v16h v; v8h h2[2]; } a0, a1;
            a0.h2[0] = *(const v8h*)(A0 + c0);
            a0.h2[1] = *(const v8h*)(A0 + c0 + 8);
            a1.h2[0] = *(const v8h*)(A1 + c0);
            a1.h2[1] = *(const v8h*)(A1 + c0 + 8);
            v16h b[4];
#pragma unroll
            for (int ni = 0; ni < 4; ++ni)
                b[ni] = *(const v16h*)(bpix[ni] + boff + c0);
#pragma unroll
            for (int ni = 0; ni < 4; ++ni) {
                acc[0][ni] = __builtin_amdgcn_wmma_f32_16x16x32_f16(
                    false, a0.v, false, b[ni], (short)0, acc[0][ni], false, false);
                acc[1][ni] = __builtin_amdgcn_wmma_f32_16x16x32_f16(
                    false, a1.v, false, b[ni], (short)0, acc[1][ni], false, false);
            }
        }
    }

    // D layout: VGPR j -> (co = cobase + mi*16 + hi*8 + j, pixel = pbase + ni*16 + l15)
#pragma unroll
    for (int mi = 0; mi < 2; ++mi) {
#pragma unroll
        for (int ni = 0; ni < 4; ++ni) {
            float* orow = out
                + ((size_t)nn[ni] * COUT + cobase + mi * 16 + hi * 8) * (H * W)
                + (size_t)hh[ni] * W + ww[ni];
#pragma unroll
            for (int j = 0; j < 8; ++j)
                orow[(size_t)j * (H * W)] = acc[mi][ni][j];
        }
    }
}

// ---------------------------------------------------------------------------
extern "C" void kernel_launch(void* const* d_in, const int* in_sizes, int n_in,
                              void* d_out, int out_size, void* d_ws, size_t ws_size,
                              hipStream_t stream) {
    const float* x   = (const float*)d_in[0];
    const float* wgt = (const float*)d_in[1];
    float*       out = (float*)d_out;

    const size_t xh_bytes = (size_t)NB * HP * WP * CIN * sizeof(hf); // 55,115,776 B
    hf* xh = (hf*)d_ws;
    hf* wb = (hf*)((char*)d_ws + xh_bytes);                          // +1,179,648 B

    binarize_w<<<(COUT * CIN * 9) / 256, 256, 0, stream>>>(wgt, wb);
    pad_nhwc  <<<(NB * HP * WP * CIN) / 256, 256, 0, stream>>>(x, xh);

    const int blocks = PIX / 64;   // 1568 blocks x 8 waves: each block = 64 pixels, all 256 co
    bconv_wmma<<<blocks, 256, 0, stream>>>(xh, wb, out);
}
